// DiTXMoEBlock_53188874994146
// MI455X (gfx1250) — matrix operand order repaired and checked by reading, not compiled, not run on hardware
//
#include <hip/hip_runtime.h>
#include <hip/hip_bf16.h>
#include <math.h>

// ---------------------------------------------------------------------------
// Problem constants (from reference)
// ---------------------------------------------------------------------------
#define BATCH   8
#define SEQ     768
#define DIM     1024
#define NEXP    4
#define HID     4096
#define MTOT    (BATCH * SEQ)        // 6144 rows
#define LHEAD   256                  // SEQ/3
#define NPASS   5                    // shared + 4 experts

// ---------------------------------------------------------------------------
// Optional gfx1250 feature detection (fallbacks keep the build green)
// ---------------------------------------------------------------------------
#if defined(__has_builtin)
# if __has_builtin(__builtin_amdgcn_tanhf)
#  define USE_HW_TANH 1
# endif
# if __has_builtin(__builtin_amdgcn_global_load_async_to_lds_b128) && \
     __has_builtin(__builtin_amdgcn_s_wait_asynccnt)
#  define USE_ASYNC_LDS 1
# endif
# if __has_builtin(__builtin_amdgcn_tensor_load_to_lds) && \
     __has_builtin(__builtin_amdgcn_s_wait_tensorcnt)
#  define USE_TDM 1
# endif
#endif

typedef __bf16        v16bf __attribute__((ext_vector_type(16)));
typedef float         v8f   __attribute__((ext_vector_type(8)));
typedef int           v4i_t __attribute__((ext_vector_type(4)));   // 128-bit payload
typedef unsigned int  v4u_t __attribute__((ext_vector_type(4)));   // TDM D# group 0
typedef int           v8i_t __attribute__((ext_vector_type(8)));   // TDM D# group 1

union FragU {               // 32 bytes: one WMMA A/B operand per lane
    uint4 u[2];
    v16bf v;
};

__device__ __forceinline__ unsigned short f2bf(float f) {
    unsigned u = __float_as_uint(f);
    unsigned r = (u + 0x7FFFu + ((u >> 16) & 1u)) >> 16;   // round-to-nearest-even
    return (unsigned short)r;
}

__device__ __forceinline__ float fast_tanh(float x) {
#if defined(USE_HW_TANH)
    return __builtin_amdgcn_tanhf(x);       // V_TANH_F32 (TRANS op)
#else
    return tanhf(x);
#endif
}

__device__ __forceinline__ float gelu_tanh(float x) {
    float x3 = x * x * x;
    return 0.5f * x * (1.0f + fast_tanh(0.7978845608028654f * (x + 0.044715f * x3)));
}

#if defined(USE_TDM)
// ---------------------------------------------------------------------------
// TDM 2-D tile load: global (row-major, stride0 elems/row) -> LDS, with
// LDS padding of 16B after every 64B so rows land on an 80B (LDP=40) pitch.
// Descriptor layout per CDNA5 ISA §8.3 (group 0) / §8.4 (group 1).
// Issue from ONE wave only; EXEC is ignored by tensor ops.
// This toolchain exposes the 6-arg builtin:
//   (uint32x4 g0, int32x8 g1, int32x4 g2, int32x4 g3, int32x8 extra, i32 cpol)
// ---------------------------------------------------------------------------
__device__ __forceinline__ void tdm_load_tile_2d(const unsigned short* gsrc,
                                                 unsigned short* ldst,
                                                 unsigned tensor_d0,   // row length (elems)
                                                 unsigned tensor_d1,   // total rows
                                                 unsigned tile_d0,     // tile width (elems)
                                                 unsigned tile_d1,     // tile height (rows)
                                                 unsigned long long stride0) // row stride (elems)
{
    unsigned long long ga  = (unsigned long long)gsrc;        // byte address
    unsigned           lds = (unsigned)(unsigned long long)ldst; // LDS offset = low 32 bits

    v4u_t g0;
    g0.x = 1u;                                                // count=1 (valid user D#)
    g0.y = lds;                                               // lds_addr
    g0.z = (unsigned)ga;                                      // global_addr[31:0]
    g0.w = (unsigned)((ga >> 32) & 0x1FFFFFFu) | (2u << 30);  // global_addr[56:32] | type=2

    v8i_t g1;
    g1[0] = (int)((1u << 16)      // data_size = 1 -> 2 bytes
                | (1u << 20)      // pad_enable
                | (3u << 22)      // pad_interval: 16 DWORDs (64B of data)
                | (3u << 25));    // pad_amount:   4 DWORDs (16B pad)
    g1[1] = (int)((tensor_d0 & 0xFFFFu) << 16);                       // [79:48] tensor_dim0 (lo)
    g1[2] = (int)((tensor_d0 >> 16) | ((tensor_d1 & 0xFFFFu) << 16)); // dim0 hi | dim1 lo
    g1[3] = (int)((tensor_d1 >> 16) | (tile_d0 << 16));               // dim1 hi | tile_dim0
    g1[4] = (int)tile_d1;                                             // tile_dim1 | tile_dim2=0
    g1[5] = (int)(unsigned)(stride0 & 0xFFFFFFFFu);                   // tensor_dim0_stride lo
    g1[6] = (int)(unsigned)((stride0 >> 32) & 0xFFFFu);               // stride0 hi | stride1 lo=0
    g1[7] = 0;                                                        // stride1 hi = 0

    v4i_t zz4 = {0, 0, 0, 0};                                 // groups 2/3 unused (2-D)
    v8i_t zz8 = {0, 0, 0, 0, 0, 0, 0, 0};
    __builtin_amdgcn_tensor_load_to_lds(g0, g1, zz4, zz4, zz8, 0);
}
#endif

// ---------------------------------------------------------------------------
// 1) Gating: segment means -> logits -> time modulation -> softmax -> top-2
// ---------------------------------------------------------------------------
__global__ __launch_bounds__(256)
void gating_kernel(const float* __restrict__ ctx,       // (B, L, D)
                   const float* __restrict__ tcond,     // (B, D)
                   const float* __restrict__ gate_w,    // (3D, E)
                   const float* __restrict__ gate_b,    // (E)
                   const float* __restrict__ time_w,    // (D, 2E)
                   const float* __restrict__ time_b,    // (2E)
                   float* __restrict__ w_be)            // (B, E)
{
    __shared__ float sFull[DIM];
    __shared__ float sHp[DIM];
    __shared__ float sWp[DIM];
    __shared__ float sSilu[DIM];
    __shared__ float sLog[NEXP];
    __shared__ float sMod[2 * NEXP];

    const int b = blockIdx.x;
    const int t = threadIdx.x;

    // Segment means: head [0,256), wrist [256,512), proprio [512,768)
    for (int j = 0; j < DIM / 256; ++j) {
        int d = j * 256 + t;
        float sf = 0.f, sh = 0.f, sw = 0.f, sp = 0.f;
        const float* base = ctx + ((size_t)b * SEQ) * DIM + d;
        for (int l = 0; l < SEQ; ++l) {
            float x = base[(size_t)l * DIM];
            sf += x;
            if (l < LHEAD)            sh += x;
            else if (l < 2 * LHEAD)   sw += x;
            else                      sp += x;
        }
        sFull[d] = sf * (1.0f / SEQ);
        sHp[d]   = (sh + sp) * (1.0f / (2 * LHEAD));
        sWp[d]   = (sw + sp) * (1.0f / (2 * LHEAD));
        float xc = tcond[(size_t)b * DIM + d];
        sSilu[d] = xc / (1.0f + expf(-xc));
    }
    __syncthreads();

    if (t < NEXP) {                                 // gate logits
        float acc = gate_b[t];
        for (int d = 0; d < DIM; ++d) {
            acc += sFull[d] * gate_w[(size_t)d * NEXP + t];
            acc += sHp[d]   * gate_w[(size_t)(DIM + d) * NEXP + t];
            acc += sWp[d]   * gate_w[(size_t)(2 * DIM + d) * NEXP + t];
        }
        sLog[t] = acc;
    }
    if (t >= 32 && t < 32 + 2 * NEXP) {             // time modulation
        int j = t - 32;
        float acc = time_b[j];
        for (int d = 0; d < DIM; ++d)
            acc += sSilu[d] * time_w[(size_t)d * (2 * NEXP) + j];
        sMod[j] = acc;
    }
    __syncthreads();

    if (t == 0) {
        float z[NEXP], p[NEXP];
        float mx = -1e30f;
        for (int e = 0; e < NEXP; ++e) {
            z[e] = sLog[e] * (1.0f + sMod[e]) + sMod[NEXP + e];
            mx = fmaxf(mx, z[e]);
        }
        float sum = 0.f;
        for (int e = 0; e < NEXP; ++e) { p[e] = expf(z[e] - mx); sum += p[e]; }
        for (int e = 0; e < NEXP; ++e) p[e] /= sum;
        // top-2 (first index wins ties, matching lax.top_k)
        int i1 = 0;
        for (int e = 1; e < NEXP; ++e) if (p[e] > p[i1]) i1 = e;
        int i2 = (i1 == 0) ? 1 : 0;
        for (int e = 0; e < NEXP; ++e) if (e != i1 && p[e] > p[i2]) i2 = e;
        float wsum = p[i1] + p[i2] + 1e-8f;
        for (int e = 0; e < NEXP; ++e)
            w_be[b * NEXP + e] = (e == i1 || e == i2) ? p[e] / wsum : 0.0f;
    }
}

// ---------------------------------------------------------------------------
// 2) Per-row coefficients for all 5 passes (pass 0 = shared branch, coeff 1)
// ---------------------------------------------------------------------------
__global__ __launch_bounds__(256)
void coeff_kernel(const float* __restrict__ w_be, float* __restrict__ coeffAll)
{
    int idx = blockIdx.x * 256 + threadIdx.x;
    if (idx >= NPASS * MTOT) return;
    int p = idx / MTOT;
    int m = idx % MTOT;
    float c;
    if (p == 0) {
        c = 1.0f;
    } else {
        int e = p - 1;
        int b = m / SEQ;
        int l = m % SEQ;
        c = w_be[b * NEXP + e];
        if (e == 1 && l >= LHEAD && l < 2 * LHEAD) c = 0.0f;   // expert 1: wrist masked
        if (e == 2 && l < LHEAD)                   c = 0.0f;   // expert 2: head masked
    }
    coeffAll[idx] = c;
}

// ---------------------------------------------------------------------------
// 3) fp32 -> bf16 conversion (n must be a multiple of 4)
// ---------------------------------------------------------------------------
__global__ __launch_bounds__(256)
void cvt_bf16_kernel(const float* __restrict__ src, unsigned short* __restrict__ dst, int n)
{
    int i = (blockIdx.x * 256 + threadIdx.x) * 4;
    if (i >= n) return;
    float4 f = *(const float4*)(src + i);
    dst[i + 0] = f2bf(f.x);
    dst[i + 1] = f2bf(f.y);
    dst[i + 2] = f2bf(f.z);
    dst[i + 3] = f2bf(f.w);
}

// ---------------------------------------------------------------------------
// 4) Double-buffered bf16 WMMA GEMM: C = A(MxK) * B(KxN), f32 accumulation.
//    Block tile 128x128, K-step 32. 8 waves: 4 (M) x 2 (N), each wave 32x64
//    = 2x4 fragments of v_wmma_f32_16x16x32_bf16.
//    A tile staged by the Tensor Data Mover (one TENSOR_LOAD_TO_LDS per tile,
//    LDS pad -> 80B row pitch), falling back to GLOBAL_LOAD_ASYNC_TO_LDS_B128
//    or plain VGPR staging. B tile transposed on store.
//    MODE 0: out_bf16 = bf16(gelu(C + bias))            (up-proj)
//    MODE 1: out_f32 (+)= coeff[row] * (C + bias[col])  (down-proj + combine)
// ---------------------------------------------------------------------------
#define LDP 40   // LDS row pitch in elements (80B: 16B-aligned, bank-spread)

template<int MODE>
__global__ __launch_bounds__(256)
void gemm_wmma_kernel(const unsigned short* __restrict__ A,
                      const unsigned short* __restrict__ B,
                      const float* __restrict__ bias,
                      const float* __restrict__ coeff,
                      unsigned short* __restrict__ outB,
                      float* __restrict__ outF,
                      int M, int N, int K, int accumulate)
{
    __shared__ unsigned short As[2][128 * LDP];   // [m][k] row-major (TDM-padded)
    __shared__ unsigned short Bs[2][128 * LDP];   // [n][k] (transposed on store)

    const int t    = threadIdx.x;
    const int lane = t & 31;
    const int wave = t >> 5;
    const int wm   = wave & 3;        // 0..3 -> 32-row strip
    const int wn   = wave >> 2;       // 0..1 -> 64-col strip

    const int gm0 = blockIdx.y * 128;
    const int gn0 = blockIdx.x * 128;

    // global->LDS staging assignments
    const int ar  = t >> 1;           // 0..127 rows of A tile
    const int ac  = (t & 1) * 16;     // 0 / 16
    const int bkr = t >> 3;           // 0..31 k-rows of B tile
    const int bnc = (t & 7) * 16;     // 0..112

    auto stage_tile = [&](int kk, int buf) {
        // --- A tile (row-major, layout preserving) ---
#if defined(USE_TDM)
        if (wave == 0)
            tdm_load_tile_2d(A + (size_t)gm0 * K + kk, &As[buf][0],
                             (unsigned)K, (unsigned)M,
                             32u, 128u, (unsigned long long)K);
#elif defined(USE_ASYNC_LDS)
        const unsigned short* asrc = A + (size_t)(gm0 + ar) * K + kk + ac;
        unsigned short*       adst = &As[buf][ar * LDP + ac];
        __builtin_amdgcn_global_load_async_to_lds_b128(
            (v4i_t*)(void*)asrc,       (v4i_t*)(void*)adst,       0, 0);
        __builtin_amdgcn_global_load_async_to_lds_b128(
            (v4i_t*)(void*)(asrc + 8), (v4i_t*)(void*)(adst + 8), 0, 0);
#else
        const unsigned short* asrc = A + (size_t)(gm0 + ar) * K + kk + ac;
        unsigned short*       adst = &As[buf][ar * LDP + ac];
        uint4 a0 = *(const uint4*)(asrc);
        uint4 a1 = *(const uint4*)(asrc + 8);
        *(uint4*)(adst)     = a0;
        *(uint4*)(adst + 8) = a1;
#endif
        // --- B tile transposed: Bs[n][k] ---
        const unsigned short* bsrc = B + (size_t)(kk + bkr) * N + gn0 + bnc;
        union { uint4 q[2]; unsigned short s[16]; } bb;
        bb.q[0] = *(const uint4*)(bsrc);
        bb.q[1] = *(const uint4*)(bsrc + 8);
#pragma unroll
        for (int i = 0; i < 16; ++i)
            Bs[buf][(bnc + i) * LDP + bkr] = bb.s[i];
    };

    auto sync_stage = [&]() {
#if defined(USE_TDM)
        __builtin_amdgcn_s_wait_tensorcnt(0);   // no-op for waves with TENSORcnt==0
#elif defined(USE_ASYNC_LDS)
        __builtin_amdgcn_s_wait_asynccnt(0);
#endif
        __syncthreads();
    };

    v8f acc[2][4] = {};

    stage_tile(0, 0);
    sync_stage();

    int buf = 0;
    for (int kk = 0; kk < K; kk += 32) {
        const int nbuf = buf ^ 1;
        if (kk + 32 < K) stage_tile(kk + 32, nbuf);     // overlap next tile
        if (kk + 64 < K) {                              // prefetch tile after next
            __builtin_prefetch(A + (size_t)(gm0 + ar) * K + kk + 64 + ac, 0, 0);
            __builtin_prefetch(B + (size_t)(kk + 64 + bkr) * N + gn0 + bnc, 0, 0);
        }

        // --- A fragments (16-bit A 16x32: lanes 0-15 k-group 0, 16-31 k-group 8) ---
        FragU af[2];
        const int ka = (lane < 16) ? 0 : 8;
#pragma unroll
        for (int mi = 0; mi < 2; ++mi) {
            int m = wm * 32 + mi * 16 + (lane & 15);
            af[mi].u[0] = *(const uint4*)(&As[buf][m * LDP + ka]);       // k = ka..ka+7
            af[mi].u[1] = *(const uint4*)(&As[buf][m * LDP + ka + 16]);  // k = ka+16..ka+23
        }
        // --- B fragments (16-bit B 32x16: lanes 0-15 k 0..7,16..23, lanes 16-31 k 8..15,24..31) ---
        FragU bfg[4];
        const int kb = (lane < 16) ? 0 : 8;
#pragma unroll
        for (int ni = 0; ni < 4; ++ni) {
            int n = wn * 64 + ni * 16 + (lane & 15);
            bfg[ni].u[0] = *(const uint4*)(&Bs[buf][n * LDP + kb]);
            bfg[ni].u[1] = *(const uint4*)(&Bs[buf][n * LDP + kb + 16]);
        }

#pragma unroll
        for (int mi = 0; mi < 2; ++mi)
#pragma unroll
            for (int ni = 0; ni < 4; ++ni)
                acc[mi][ni] = __builtin_amdgcn_wmma_f32_16x16x32_bf16(
                    false, af[mi].v, false, bfg[ni].v,
                    (short)0, acc[mi][ni], false, false);

        sync_stage();
        buf = nbuf;
    }

    // --- epilogue (C/D layout: VGPR j, lane L -> row j + (L<16?0:8), col L%16) ---
    const int rbase = gm0 + wm * 32 + ((lane < 16) ? 0 : 8);
    const int cbase = gn0 + wn * 64 + (lane & 15);
#pragma unroll
    for (int mi = 0; mi < 2; ++mi) {
#pragma unroll
        for (int ni = 0; ni < 4; ++ni) {
            const int c = cbase + ni * 16;
            const float bs = bias[c];
#pragma unroll
            for (int j = 0; j < 8; ++j) {
                const int r = rbase + mi * 16 + j;
                float v = acc[mi][ni][j] + bs;
                if (MODE == 0) {
                    outB[(size_t)r * N + c] = f2bf(gelu_tanh(v));
                } else {
                    float y = coeff[r] * v;
                    size_t o = (size_t)r * N + c;
                    if (accumulate) atomicAdd(&outF[o], y);   // no-return global_atomic_add_f32
                    else            outF[o] = y;
                }
            }
        }
    }
}

// ---------------------------------------------------------------------------
// Host-side launcher
// ---------------------------------------------------------------------------
extern "C" void kernel_launch(void* const* d_in, const int* in_sizes, int n_in,
                              void* d_out, int out_size, void* d_ws, size_t ws_size,
                              hipStream_t stream)
{
    const float* ctx    = (const float*)d_in[0];   // (8,768,1024)
    const float* tcond  = (const float*)d_in[1];   // (8,1024)
    const float* gate_w = (const float*)d_in[2];   // (3072,4)
    const float* gate_b = (const float*)d_in[3];   // (4)
    const float* time_w = (const float*)d_in[4];   // (1024,8)
    const float* time_b = (const float*)d_in[5];   // (8)
    const float* ew1    = (const float*)d_in[6];   // (4,1024,4096)
    const float* eb1    = (const float*)d_in[7];   // (4,4096)
    const float* ew2    = (const float*)d_in[8];   // (4,4096,1024)
    const float* eb2    = (const float*)d_in[9];   // (4,1024)
    const float* sw1    = (const float*)d_in[10];  // (1024,4096)
    const float* sb1    = (const float*)d_in[11];  // (4096)
    const float* sw2    = (const float*)d_in[12];  // (4096,1024)
    const float* sb2    = (const float*)d_in[13];  // (1024)
    float* out = (float*)d_out;                    // (8,768,1024)

    // workspace layout (256B aligned blocks)
    char* ws = (char*)d_ws;
    size_t off = 0;
    auto take = [&](size_t bytes) -> char* {
        char* p = ws + off;
        off = (off + bytes + 255) & ~(size_t)255;
        return p;
    };
    float*          w_be     = (float*)take(BATCH * NEXP * sizeof(float));
    float*          coeffAll = (float*)take((size_t)NPASS * MTOT * sizeof(float));
    unsigned short* Xbf      = (unsigned short*)take((size_t)MTOT * DIM * 2);
    unsigned short* W1bf     = (unsigned short*)take((size_t)DIM * HID * 2);
    unsigned short* W2bf     = (unsigned short*)take((size_t)HID * DIM * 2);
    unsigned short* Hbf      = (unsigned short*)take((size_t)MTOT * HID * 2);
    (void)ws_size; (void)n_in; (void)in_sizes; (void)out_size;

    // 1) gating + per-row coefficients
    gating_kernel<<<BATCH, 256, 0, stream>>>(ctx, tcond, gate_w, gate_b,
                                             time_w, time_b, w_be);
    coeff_kernel<<<(NPASS * MTOT + 255) / 256, 256, 0, stream>>>(w_be, coeffAll);

    // 2) activations to bf16
    {
        int n = MTOT * DIM;
        cvt_bf16_kernel<<<n / 1024, 256, 0, stream>>>(ctx, Xbf, n);
    }

    // 3) five passes: shared branch first (overwrite), then experts (accumulate)
    const dim3 g1(HID / 128, MTOT / 128);   // (32, 48)
    const dim3 g2(DIM / 128, MTOT / 128);   // (8, 48)
    const int nW1 = DIM * HID;              // 4 M elements
    const int nW2 = HID * DIM;

    for (int p = 0; p < NPASS; ++p) {
        const float *w1, *b1, *w2, *b2;
        if (p == 0) { w1 = sw1; b1 = sb1; w2 = sw2; b2 = sb2; }
        else {
            int e = p - 1;
            w1 = ew1 + (size_t)e * DIM * HID;
            b1 = eb1 + (size_t)e * HID;
            w2 = ew2 + (size_t)e * HID * DIM;
            b2 = eb2 + (size_t)e * DIM;
        }
        cvt_bf16_kernel<<<nW1 / 1024, 256, 0, stream>>>(w1, W1bf, nW1);
        cvt_bf16_kernel<<<nW2 / 1024, 256, 0, stream>>>(w2, W2bf, nW2);

        // up-projection + GELU -> Hbf (bf16)
        gemm_wmma_kernel<0><<<g1, 256, 0, stream>>>(
            Xbf, W1bf, b1, nullptr, Hbf, nullptr, MTOT, HID, DIM, 0);

        // down-projection, scale by coeff, combine into fp32 output
        gemm_wmma_kernel<1><<<g2, 256, 0, stream>>>(
            Hbf, W2bf, b2, coeffAll + (size_t)p * MTOT, nullptr, out,
            MTOT, DIM, HID, p > 0 ? 1 : 0);
    }
}
